// TFLMGraphConv_3908420240215
// MI455X (gfx1250) — matrix operand order repaired and checked by reading, not compiled, not run on hardware
//
#include <hip/hip_runtime.h>
#include <math.h>

#define Bn 2
#define Cin 35
#define Tt 8
#define Nn 1024
#define Fc 32
#define Dk 16
#define Cout 64

typedef __attribute__((ext_vector_type(16))) _Float16 v16h;
typedef __attribute__((ext_vector_type(8)))  float    v8f;
typedef __attribute__((ext_vector_type(2)))  float    v2f;

// ---------------------------------------------------------------------------
// Stage 0: per-node prep. pe = Wpe*p + b; Xp=[feat,pe]; Q=(Wq*Xp)/sqrt(16) f16,
// K=Wk*Xp f16, V=Wv*feat f32, u = ca_w1*p (for pairwise gate). One thread/node.
// ---------------------------------------------------------------------------
__global__ void prep_kernel(const float* __restrict__ x,
                            const float* __restrict__ pew, const float* __restrict__ peb,
                            const float* __restrict__ wq,  const float* __restrict__ wk,
                            const float* __restrict__ wv,  const float* __restrict__ caw1,
                            _Float16* __restrict__ Qh, _Float16* __restrict__ Kh,
                            float* __restrict__ Vf, float* __restrict__ U) {
  int idx = blockIdx.x * blockDim.x + threadIdx.x;   // < B*T*N
  int n = idx & (Nn - 1);
  int t = (idx >> 10) & 7;
  int b = idx >> 13;
  const size_t cs = (size_t)Tt * Nn;                 // channel stride in x
  size_t xoff = (size_t)b * Cin * cs + (size_t)t * Nn + n;
  float p[3];
  for (int c = 0; c < 3; c++) p[c] = x[xoff + (size_t)c * cs];
  float f[Fc];
  for (int c = 0; c < Fc; c++) f[c] = x[xoff + (size_t)(3 + c) * cs];
  float pe[8];
  for (int o = 0; o < 8; o++) {
    float a = peb[o];
    for (int c = 0; c < 3; c++) a += pew[o * 3 + c] * p[c];
    pe[o] = a;
  }
  size_t nb = (size_t)(b * Tt + t) * Nn + n;
  for (int d = 0; d < Dk; d++) {
    float q = 0.f, k = 0.f, v = 0.f;
    for (int c = 0; c < Fc; c++) {
      q += wq[d * 40 + c] * f[c];
      k += wk[d * 40 + c] * f[c];
      v += wv[d * 32 + c] * f[c];
    }
    for (int c = 0; c < 8; c++) {
      q += wq[d * 40 + 32 + c] * pe[c];
      k += wk[d * 40 + 32 + c] * pe[c];
    }
    Qh[nb * 16 + d] = (_Float16)(q * 0.25f);   // fold 1/sqrt(d_k)=1/4 into Q
    Kh[nb * 16 + d] = (_Float16)k;
    Vf[nb * 16 + d] = v;
  }
  for (int o = 0; o < 8; o++) {
    float a = 0.f;
    for (int c = 0; c < 3; c++) a += caw1[o * 3 + c] * p[c];
    U[nb * 8 + o] = a;
  }
}

// ---------------------------------------------------------------------------
// Stage 1: S = Q * K^T (scale pre-folded). One wave per 16x16 tile.
// v_wmma_f32_16x16x32_f16 with K=16 real, K=16..31 zero padding.
// A frag: lanes 0-15 -> row, elems 0..15 hold K0..15; lanes 16-31 are zero pad.
// B frag (=K^T): lanes 0-15 hold K0..15 of their column; lanes 16-31 are zero pad.
// ---------------------------------------------------------------------------
__global__ void score_kernel(const _Float16* __restrict__ Qh,
                             const _Float16* __restrict__ Kh,
                             float* __restrict__ S, int t) {
  int i = blockIdx.x, j = blockIdx.y, b = blockIdx.z;
  int lane = threadIdx.x;
  int hi = lane >> 4, lo = lane & 15;
  v16h a = {};
  v16h bb = {};
  {
    size_t qrow = ((size_t)(b * Tt + t) * Nn + i * 16 + lo) * 16;
    if (hi == 0)                            // lanes0-15 carry K=0..15 in elems 0..15
      for (int k = 0; k < 16; k++) a[k] = Qh[qrow + k];
    // lanes 16-31 remain zero (K=16..31 pad)
  }
  if (hi == 0) {
    size_t krow = ((size_t)(b * Tt + t) * Nn + j * 16 + lo) * 16;
    for (int k = 0; k < 16; k++) bb[k] = Kh[krow + k];
  }
  v8f c = {};
  c = __builtin_amdgcn_wmma_f32_16x16x32_f16(false, a, false, bb, (short)0, c,
                                             false, false);
  size_t base = ((size_t)b * Nn + (size_t)i * 16) * Nn + (size_t)j * 16;
  for (int r = 0; r < 8; r++)
    S[base + (size_t)(r + 8 * hi) * Nn + lo] = c[r];   // C layout: (M=r+8*hi, N=lo)
}

// ---------------------------------------------------------------------------
// Stage 2: pairwise positional gate. att(n,m) from u_n - u_m (linearity of the
// 3->8 conv over the position difference). S *= att. One thread per element.
// ---------------------------------------------------------------------------
__global__ void attmul_kernel(const float* __restrict__ U, float* __restrict__ S,
                              const float* __restrict__ cb1, const float* __restrict__ cg,
                              const float* __restrict__ cbt, const float* __restrict__ cw2,
                              const float* __restrict__ cb2, int t) {
  int e = blockIdx.x * blockDim.x + threadIdx.x;   // < N*N
  int b = blockIdx.y;
  int n = e >> 10, m = e & 1023;
  const float inv_std = rsqrtf(1.0f + 1e-5f);
  size_t ub = (size_t)(b * Tt + t) * Nn * 8;
  const float* un = U + ub + (size_t)n * 8;
  const float* um = U + ub + (size_t)m * 8;
  float acc = 0.f;
  for (int o = 0; o < 8; o++) {
    float h  = (un[o] - um[o] + cb1[o]) * inv_std * cg[o] + cbt[o];
    float hs = h * fminf(fmaxf(h + 3.f, 0.f), 6.f) * (1.f / 6.f);   // h_swish
    acc += hs * cw2[o];
  }
  float att = fminf(fmaxf(acc + cb2[0] + 3.f, 0.f), 6.f) * (1.f / 6.f); // h_sig
  S[((size_t)b * Nn + n) * Nn + m] *= att;
}

// ---------------------------------------------------------------------------
// Stage 3: row softmax. One 256-thread block per row; LDS tree reductions.
// ---------------------------------------------------------------------------
__global__ void softmax_kernel(const float* __restrict__ S, float* __restrict__ A) {
  int row = blockIdx.x, b = blockIdx.y, tid = threadIdx.x;
  size_t base = ((size_t)b * Nn + row) * Nn;
  __shared__ float red[256];
  float v[4];
  for (int k = 0; k < 4; k++) v[k] = S[base + tid + k * 256];
  float mx = fmaxf(fmaxf(v[0], v[1]), fmaxf(v[2], v[3]));
  red[tid] = mx; __syncthreads();
  for (int s = 128; s > 0; s >>= 1) {
    if (tid < s) red[tid] = fmaxf(red[tid], red[tid + s]);
    __syncthreads();
  }
  mx = red[0]; __syncthreads();
  float e[4], sum = 0.f;
  for (int k = 0; k < 4; k++) { e[k] = __expf(v[k] - mx); sum += e[k]; }
  red[tid] = sum; __syncthreads();
  for (int s = 128; s > 0; s >>= 1) {
    if (tid < s) red[tid] += red[tid + s];
    __syncthreads();
  }
  float rinv = 1.f / red[0];
  for (int k = 0; k < 4; k++) A[base + tid + k * 256] = e[k] * rinv;
}

// ---------------------------------------------------------------------------
// Stage 4: symmetrize: S = 0.5*(A + A^T)
// ---------------------------------------------------------------------------
__global__ void sym_kernel(const float* __restrict__ A, float* __restrict__ S) {
  int e = blockIdx.x * blockDim.x + threadIdx.x;
  int b = blockIdx.y;
  int n = e >> 10, m = e & 1023;
  size_t bb = (size_t)b * Nn * Nn;
  S[bb + (size_t)n * Nn + m] =
      0.5f * (A[bb + (size_t)n * Nn + m] + A[bb + (size_t)m * Nn + n]);
}

// ---------------------------------------------------------------------------
// Stage 5: per-row top-4 (iterative argmax, lower-index tiebreak), mask,
// renorm (+1e-8), EMA with beta (t==0 takes the fresh A). Writes dense A_prev.
// ---------------------------------------------------------------------------
__global__ void topk_ema_kernel(const float* __restrict__ Asym,
                                float* __restrict__ Aprev,
                                const float* __restrict__ beta_p, int t) {
  int row = blockIdx.x, b = blockIdx.y, tid = threadIdx.x;
  size_t base = ((size_t)b * Nn + row) * Nn;
  __shared__ float sv[256];
  __shared__ int   si[256];
  __shared__ int   cidx[4];
  __shared__ float cval[4];
  float v[4]; int mm[4];
  for (int k = 0; k < 4; k++) { mm[k] = tid + k * 256; v[k] = Asym[base + mm[k]]; }
  for (int pass = 0; pass < 4; pass++) {
    float bv = -3.4e38f; int bi = Nn;
    for (int k = 0; k < 4; k++) {
      bool taken = false;
      for (int p = 0; p < pass; p++) taken = taken || (mm[k] == cidx[p]);
      if (!taken && (v[k] > bv || (v[k] == bv && mm[k] < bi))) { bv = v[k]; bi = mm[k]; }
    }
    sv[tid] = bv; si[tid] = bi; __syncthreads();
    for (int s = 128; s > 0; s >>= 1) {
      if (tid < s) {
        float ov = sv[tid + s]; int oi = si[tid + s];
        if (ov > sv[tid] || (ov == sv[tid] && oi < si[tid])) { sv[tid] = ov; si[tid] = oi; }
      }
      __syncthreads();
    }
    if (tid == 0) { cidx[pass] = si[0]; cval[pass] = sv[0]; }
    __syncthreads();
  }
  float inv = 1.f / (cval[0] + cval[1] + cval[2] + cval[3] + 1e-8f);
  float bet = beta_p[0];
  for (int k = 0; k < 4; k++) {
    bool sel = false;
    for (int p = 0; p < 4; p++) sel = sel || (mm[k] == cidx[p]);
    float g = sel ? v[k] * inv : 0.f;
    float o = (t == 0) ? g : bet * Aprev[base + mm[k]] + (1.f - bet) * g;
    Aprev[base + mm[k]] = o;
  }
}

// ---------------------------------------------------------------------------
// Stage 6: Hav = A_t * V  (1024x1024x16, f32 WMMA 16x16x4, K-loop of 256).
// One wave per 16-row stripe; N-tile = full d_k = 16.
// ---------------------------------------------------------------------------
__global__ void av_kernel(const float* __restrict__ A, const float* __restrict__ Vf,
                          float* __restrict__ Hav, int t) {
  int i = blockIdx.x, b = blockIdx.y;
  int lane = threadIdx.x, hi = lane >> 4, lo = lane & 15;
  size_t abase = ((size_t)b * Nn + i * 16 + lo) * Nn;
  size_t vbase = (size_t)(b * Tt + t) * Nn * 16;
#if __has_builtin(__builtin_amdgcn_wmma_f32_16x16x4_f32)
  v8f acc = {};
  for (int k0 = 0; k0 < Nn; k0 += 4) {
    v2f av, bv;
    // A frag (16x4 f32): lanes 0-15 -> K=k0,k0+1 ; lanes 16-31 -> K=k0+2,k0+3
    av[0] = A[abase + k0 + 2 * hi + 0];
    av[1] = A[abase + k0 + 2 * hi + 1];
    // B frag (4x16 f32): lane = column; same K split across lane halves
    bv[0] = Vf[vbase + (size_t)(k0 + 2 * hi + 0) * 16 + lo];
    bv[1] = Vf[vbase + (size_t)(k0 + 2 * hi + 1) * 16 + lo];
    acc = __builtin_amdgcn_wmma_f32_16x16x4_f32(false, av, false, bv, (short)0,
                                                acc, false, false);
  }
  size_t hb = ((size_t)b * Nn + (size_t)i * 16) * 16;
  for (int r = 0; r < 8; r++)
    Hav[hb + (size_t)(r + 8 * hi) * 16 + lo] = acc[r];
#else
  // scalar fallback: each lane covers rows (lane) of stripe? keep it simple:
  // lane handles row (i*16 + (lane&15)), half of d_k per lane-half.
  int rrow = i * 16 + lo;
  size_t ab = ((size_t)b * Nn + rrow) * Nn;
  for (int d = hi * 8; d < hi * 8 + 8; d++) {
    float s = 0.f;
    for (int m = 0; m < Nn; m++) s += A[ab + m] * Vf[vbase + (size_t)m * 16 + d];
    Hav[((size_t)b * Nn + rrow) * 16 + d] = s;
  }
#endif
}

// ---------------------------------------------------------------------------
// Stage 7: out[b,c,t,n] = Hav[b,n,:] . weight[:,c] + bias[c]
// ---------------------------------------------------------------------------
__global__ void outproj_kernel(const float* __restrict__ Hav,
                               const float* __restrict__ W,
                               const float* __restrict__ bias,
                               float* __restrict__ out, int t) {
  int e = blockIdx.x * blockDim.x + threadIdx.x;   // < N*Cout
  int b = blockIdx.y;
  int c = e & 63, n = e >> 6;
  const float* h = Hav + ((size_t)b * Nn + n) * 16;
  float a = bias[c];
  for (int d = 0; d < 16; d++) a += h[d] * W[d * 64 + c];
  out[(((size_t)b * Cout + c) * Tt + t) * Nn + n] = a;
}

// ---------------------------------------------------------------------------
extern "C" void kernel_launch(void* const* d_in, const int* in_sizes, int n_in,
                              void* d_out, int out_size, void* d_ws, size_t ws_size,
                              hipStream_t stream) {
  (void)in_sizes; (void)n_in; (void)out_size; (void)ws_size;
  const float* x      = (const float*)d_in[0];
  const float* beta   = (const float*)d_in[1];
  const float* pew    = (const float*)d_in[2];
  const float* peb    = (const float*)d_in[3];
  // d_in[4], d_in[5]: pos_conv_* (dead code in reference)
  const float* wq     = (const float*)d_in[6];
  const float* wk     = (const float*)d_in[7];
  const float* wv     = (const float*)d_in[8];
  const float* caw1   = (const float*)d_in[9];
  const float* cab1   = (const float*)d_in[10];
  const float* cag    = (const float*)d_in[11];
  const float* cabeta = (const float*)d_in[12];
  const float* caw2   = (const float*)d_in[13];
  const float* cab2   = (const float*)d_in[14];
  const float* W      = (const float*)d_in[15];
  const float* bias   = (const float*)d_in[16];
  float* out = (float*)d_out;

  char* w = (char*)d_ws;
  auto take = [&](size_t bytes) {
    char* p = w;
    w += (bytes + 255) & ~(size_t)255;
    return p;
  };
  _Float16* Qh = (_Float16*)take((size_t)Bn * Tt * Nn * 16 * 2);
  _Float16* Kh = (_Float16*)take((size_t)Bn * Tt * Nn * 16 * 2);
  float* Vf  = (float*)take((size_t)Bn * Tt * Nn * 16 * 4);
  float* U   = (float*)take((size_t)Bn * Tt * Nn * 8 * 4);
  float* S   = (float*)take((size_t)Bn * Nn * Nn * 4);
  float* Aa  = (float*)take((size_t)Bn * Nn * Nn * 4);
  float* Ap  = (float*)take((size_t)Bn * Nn * Nn * 4);
  float* Hav = (float*)take((size_t)Bn * Nn * 16 * 4);

  prep_kernel<<<dim3((Bn * Tt * Nn) / 256), 256, 0, stream>>>(
      x, pew, peb, wq, wk, wv, caw1, Qh, Kh, Vf, U);

  for (int t = 0; t < Tt; t++) {
    score_kernel<<<dim3(Nn / 16, Nn / 16, Bn), 32, 0, stream>>>(Qh, Kh, S, t);
    attmul_kernel<<<dim3((Nn * Nn) / 256, Bn), 256, 0, stream>>>(
        U, S, cab1, cag, cabeta, caw2, cab2, t);
    softmax_kernel<<<dim3(Nn, Bn), 256, 0, stream>>>(S, Aa);
    sym_kernel<<<dim3((Nn * Nn) / 256, Bn), 256, 0, stream>>>(Aa, S);
    topk_ema_kernel<<<dim3(Nn, Bn), 256, 0, stream>>>(S, Ap, beta, t);
    av_kernel<<<dim3(Nn / 16, Bn), 32, 0, stream>>>(Ap, Vf, Hav, t);
    outproj_kernel<<<dim3((Nn * Cout) / 256, Bn), 256, 0, stream>>>(
        Hav, W, bias, out, t);
  }
}
